// SparseWindowAttention_14070312862362
// MI455X (gfx1250) — matrix operand order repaired and checked by reading, compiled, not run on hardware
//
#include <hip/hip_runtime.h>
#include <hip/hip_bf16.h>

// ---------------------------------------------------------------------------
// Types and WMMA fragment helpers (CDNA5 / gfx1250, wave32)
// ---------------------------------------------------------------------------
typedef __attribute__((ext_vector_type(16))) _Float16 v16h;
typedef __attribute__((ext_vector_type(8)))  float    v8f;

union F16x16 { v16h v; uint4 u[2]; _Float16 h[16]; };

// A-matrix (16x32 f16) fragment: lane l -> row = (l&15); halves 0..7 cover
// K = g*8..g*8+7, halves 8..15 cover K = 16+g*8..16+g*8+7  (g = l>>4).
// rowp is the per-lane row base pointer; chunks are 16B each.
__device__ __forceinline__ v16h ldA(const _Float16* rowp, int k0, int g) {
    F16x16 f;
    const _Float16* p = rowp + k0 + g * 8;
    f.u[0] = *(const uint4*)(p);
    f.u[1] = *(const uint4*)(p + 16);
    return f.v;
}

// B-matrix (32x16 f16) fragment from row-major R[N][K] (k fastest):
// lane l -> col n = (l&15); halves 0..7 cover K = k0 + g*8 .. k0 + g*8 + 7,
// halves 8..15 cover K = k0 + 16 + g*8 .. +7. rowp is the per-lane row base (row n).
__device__ __forceinline__ v16h ldB(const _Float16* rowp, int k0, int g) {
    F16x16 f;
    const _Float16* p = rowp + k0 + g * 8;
    f.u[0] = *(const uint4*)(p);
    f.u[1] = *(const uint4*)(p + 16);
    return f.v;
}

__device__ __forceinline__ v8f wmma_f16(v16h a, v16h b, v8f c) {
    return __builtin_amdgcn_wmma_f32_16x16x32_f16(
        /*neg_a=*/false, a, /*neg_b=*/false, b,
        /*c_mod=*/(short)0, c, /*reuse_a=*/false, /*reuse_b=*/false);
}

// ---------------------------------------------------------------------------
// f32 -> f16 conversion
// ---------------------------------------------------------------------------
__global__ void cvt_f32_f16(const float* __restrict__ in, _Float16* __restrict__ out, int n) {
    int i = blockIdx.x * 256 + threadIdx.x;
    if (i < n) out[i] = (_Float16)in[i];
}

// ---------------------------------------------------------------------------
// GEMM: C[M,N] = A[M,K] * W[N,K]^T + bias[N]
// A, W in f16 row-major; C in OutT. Block = 256 threads (8 waves),
// block tile 128(M) x 128(N); wave tile 16(M) x 128(N); K step 32.
// W tile (128x32 = 8KB) staged in LDS once per K-step, double-buffered:
// global loads for step i+1 overlap WMMAs of step i. Grid: (M/128, N/128).
// ---------------------------------------------------------------------------
template <typename OutT>
__global__ __launch_bounds__(256)
void gemm_nt(const _Float16* __restrict__ A, const _Float16* __restrict__ W,
             const float* __restrict__ bias, OutT* __restrict__ C,
             int M, int N, int K)
{
    __shared__ _Float16 Wlds[2][128][48];   // row stride 96B (16B aligned)

    const int wv = threadIdx.x >> 5;
    const int l  = threadIdx.x & 31;
    const int ln = l & 15;
    const int g  = l >> 4;

    const int row0 = blockIdx.x * 128 + wv * 16;
    const int col0 = blockIdx.y * 128;

    const _Float16* arow = A + (size_t)(row0 + ln) * K;

    // staging role: 2 threads per W row; each stages 32B (16 halves)
    const int srow  = threadIdx.x >> 1;
    const int spart = (threadIdx.x & 1) * 16;
    const _Float16* wrow = W + (size_t)(col0 + srow) * K + spart;

    v8f vz = {0.f, 0.f, 0.f, 0.f, 0.f, 0.f, 0.f, 0.f};
    v8f acc[8];
#pragma unroll
    for (int nt = 0; nt < 8; ++nt) acc[nt] = vz;

    // prologue: stage k0 = 0 into buffer 0
    {
        uint4 w0 = *(const uint4*)(wrow);
        uint4 w1 = *(const uint4*)(wrow + 8);
        *(uint4*)&Wlds[0][srow][spart]     = w0;
        *(uint4*)&Wlds[0][srow][spart + 8] = w1;
    }

    const int nk = K >> 5;
    for (int i = 0; i < nk; ++i) {
        __syncthreads();   // staging of buf (i&1) visible; reads of buf ((i+1)&1) done
        const int k0 = i << 5;

        if (i + 1 < nk) {  // stage next tile while computing this one
            const _Float16* wsrc = wrow + k0 + 32;
            uint4 w0 = *(const uint4*)(wsrc);
            uint4 w1 = *(const uint4*)(wsrc + 8);
            const int nb = (i + 1) & 1;
            *(uint4*)&Wlds[nb][srow][spart]     = w0;
            *(uint4*)&Wlds[nb][srow][spart + 8] = w1;
        }

        v16h a = ldA(arow, k0, g);
        if (k0 + 64 < K) __builtin_prefetch(arow + k0 + 64, 0, 0);

        const int cb = i & 1;
#pragma unroll
        for (int nt = 0; nt < 8; ++nt) {
            v16h b = ldB(&Wlds[cb][nt * 16 + ln][0], 0, g);
            acc[nt] = wmma_f16(a, b, acc[nt]);
        }
    }

#pragma unroll
    for (int nt = 0; nt < 8; ++nt) {
        const int col = col0 + nt * 16 + ln;
        const float bv = bias[col];
#pragma unroll
        for (int r = 0; r < 8; ++r) {
            C[(size_t)(row0 + g * 8 + r) * N + col] = (OutT)(acc[nt][r] + bv);
        }
    }
}

// ---------------------------------------------------------------------------
// Depthwise 4x4 pooling: px16[b,t,8,8,1024] = sum_{dy,dx} pool_w*x + pool_b
// ---------------------------------------------------------------------------
__global__ void pool_kernel(const float* __restrict__ x, const float* __restrict__ pw,
                            const float* __restrict__ pb, _Float16* __restrict__ px16)
{
    int idx = blockIdx.x * 256 + threadIdx.x;   // 2*8*8*8*1024 = 1048576 total
    if (idx >= 2 * 8 * 64 * 1024) return;
    int c = idx & 1023;
    int u = idx >> 10;
    int pxn = u & 7; u >>= 3;
    int pyn = u & 7; u >>= 3;
    int t = u & 7;
    int bb = u >> 3;
    float acc = pb[c];
#pragma unroll
    for (int dy = 0; dy < 4; ++dy)
#pragma unroll
        for (int dx = 0; dx < 4; ++dx) {
            float wv = pw[(dy * 4 + dx) * 1024 + c];
            float xv = x[((((size_t)bb * 8 + t) * 32 + pyn * 4 + dy) * 32 + (pxn * 4 + dx)) * 1024 + c];
            acc += wv * xv;
        }
    px16[idx] = (_Float16)acc;
}

// ---------------------------------------------------------------------------
// Fused sparse-window attention. One workgroup (256 thr, 8 waves) per
// (b, group n in [0,64), head h). Group n: window w = n>>2 (wr=w>>2, wc=w&3),
// time pair tpair = n&3 (t = tpair*2 + tt). 128 queries, 640 keys.
// Key slot s in [0,320) per tt: [0,64) window, [64,256) rolled valid,
// [256,320) pooled. Flash-style online softmax over 10 blocks of 64 keys.
// ---------------------------------------------------------------------------
__global__ __launch_bounds__(256)
void attn_kernel(const _Float16* __restrict__ q16, const _Float16* __restrict__ k16,
                 const _Float16* __restrict__ v16, const _Float16* __restrict__ pk16,
                 const _Float16* __restrict__ pv16, _Float16* __restrict__ y16)
{
    __shared__ _Float16 Klds[64][136];      // [key][ch], padded rows (272B)
    __shared__ _Float16 Vtlds[128][72];     // [ch][key], padded rows (144B)
    __shared__ _Float16 Plds[8][16][72];    // per-wave P tile [query][key]

    const int bid = blockIdx.x;
    const int b = bid >> 9;
    const int rem = bid & 511;
    const int n = rem >> 3;
    const int h = rem & 7;
    const int w = n >> 2, tpair = n & 3;
    const int wr = w >> 2, wc = w & 3;

    const int tid = threadIdx.x;
    const int wv = tid >> 5;
    const int l = tid & 31;
    const int ln = l & 15;
    const int g = l >> 4;

    // ---- Q A-fragments: per-lane query row (A-frag row = ln) ----
    const int qq = wv * 16 + ln;               // query index 0..127
    const int ttq = qq >> 6, pq = qq & 63;
    const int tiq = tpair * 2 + ttq;
    const int yq = wr * 8 + (pq >> 3), xq = wc * 8 + (pq & 7);
    const _Float16* qrow =
        q16 + ((((size_t)b * 8 + tiq) * 32 + yq) * 32 + xq) * 1024 + h * 128;

    v16h qa[4];
#pragma unroll
    for (int kc = 0; kc < 4; ++kc) qa[kc] = ldA(qrow, kc * 32, g);

    v8f vz = {0.f, 0.f, 0.f, 0.f, 0.f, 0.f, 0.f, 0.f};
    v8f o[8];
#pragma unroll
    for (int nt = 0; nt < 8; ++nt) o[nt] = vz;
    float mr[8], lr[8];
#pragma unroll
    for (int r = 0; r < 8; ++r) { mr[r] = -3.0e38f; lr[r] = 0.f; }

    const float scale = 0.08838834764831845f;  // 1/sqrt(128)

    // gather roles: 4 threads per key, 32 channels each
    const int kk_g = tid >> 2;
    const int c0 = (tid & 3) * 32;

    for (int blk = 0; blk < 10; ++blk) {
        __syncthreads();   // previous block's LDS reads are done

        // ---------------- gather 64 keys into LDS ----------------
        {
            const int jg = blk * 64 + kk_g;
            const int tt = (jg >= 320) ? 1 : 0;
            const int s = jg - tt * 320;
            const int t = tpair * 2 + tt;
            size_t off = 0;
            bool valid = true;
            bool pooled = false;
            if (s < 64) {
                const int y = wr * 8 + (s >> 3), x = wc * 8 + (s & 7);
                off = ((((size_t)b * 8 + t) * 32 + y) * 32 + x) * 1024 + h * 128;
            } else if (s < 256) {
                const int sr = s - 64;
                const int si = sr / 48;
                const int vi = sr - si * 48;
                const int invR = (si & 2) ? 4 : 0;
                const int invC = (si & 1) ? 4 : 0;
                int i, jx;
                if (invR == 0) {
                    if (vi < 16) { i = vi >> 2; jx = (4 - invC) + (vi & 3); }
                    else         { int u = vi - 16; i = 4 + (u >> 3); jx = u & 7; }
                } else {
                    if (vi < 32) { i = vi >> 3; jx = vi & 7; }
                    else         { int u = vi - 32; i = 4 + (u >> 2); jx = (4 - invC) + (u & 3); }
                }
                const int sh = (si & 2) ? 4 : -4;
                const int sw = (si & 1) ? 4 : -4;
                const int y = wr * 8 + i, x = wc * 8 + jx;
                const bool rok = (sh > 0) ? (y >= sh) : (y < 32 + sh);
                const bool cok = (sw > 0) ? (x >= sw) : (x < 32 + sw);
                valid = rok && cok;
                const int ys = (y - sh) & 31, xs = (x - sw) & 31;
                off = ((((size_t)b * 8 + t) * 32 + ys) * 32 + xs) * 1024 + h * 128;
            } else {
                pooled = true;
                const int pp = s - 256;
                off = (((size_t)b * 8 + t) * 64 + pp) * 1024 + h * 128;
            }
            const _Float16* ksrc = (pooled ? pk16 : k16) + off + c0;
            const _Float16* vsrc = (pooled ? pv16 : v16) + off + c0;

            uint4 z4; z4.x = z4.y = z4.z = z4.w = 0u;
            F16x16 kf[2], vf[2];
            kf[0].u[0] = kf[0].u[1] = kf[1].u[0] = kf[1].u[1] = z4;
            vf[0].u[0] = vf[0].u[1] = vf[1].u[0] = vf[1].u[1] = z4;
            if (valid) {
#pragma unroll
                for (int part = 0; part < 2; ++part) {
                    kf[part].u[0] = *(const uint4*)(ksrc + part * 16);
                    kf[part].u[1] = *(const uint4*)(ksrc + part * 16 + 8);
                    vf[part].u[0] = *(const uint4*)(vsrc + part * 16);
                    vf[part].u[1] = *(const uint4*)(vsrc + part * 16 + 8);
                }
            }
#pragma unroll
            for (int part = 0; part < 2; ++part) {
                *(uint4*)&Klds[kk_g][c0 + part * 16]     = kf[part].u[0];
                *(uint4*)&Klds[kk_g][c0 + part * 16 + 8] = kf[part].u[1];
#pragma unroll
                for (int cc = 0; cc < 16; ++cc)
                    Vtlds[c0 + part * 16 + cc][kk_g] = vf[part].h[cc];
            }
        }
        __syncthreads();

        // ---------------- S = Q * K^T (16 queries x 64 keys per wave) -------
        v8f sfr[4];
#pragma unroll
        for (int jt = 0; jt < 4; ++jt) sfr[jt] = vz;
#pragma unroll
        for (int kc = 0; kc < 4; ++kc) {
#pragma unroll
            for (int jt = 0; jt < 4; ++jt) {
                v16h bf = ldB(&Klds[jt * 16 + ln][0], kc * 32, g);
                sfr[jt] = wmma_f16(qa[kc], bf, sfr[jt]);
            }
        }

        // ---------------- online softmax update ----------------------------
        float nm[8], al[8];
#pragma unroll
        for (int r = 0; r < 8; ++r) {
            float vm = sfr[0][r];
            vm = fmaxf(vm, sfr[1][r]);
            vm = fmaxf(vm, sfr[2][r]);
            vm = fmaxf(vm, sfr[3][r]);
            vm *= scale;
#pragma unroll
            for (int msk = 1; msk < 16; msk <<= 1)
                vm = fmaxf(vm, __shfl_xor(vm, msk, 32));
            nm[r] = fmaxf(mr[r], vm);
            al[r] = __expf(mr[r] - nm[r]);
        }
#pragma unroll
        for (int r = 0; r < 8; ++r) {
            float rowsum = 0.f;
#pragma unroll
            for (int jt = 0; jt < 4; ++jt) {
                float pe = __expf(sfr[jt][r] * scale - nm[r]);
                rowsum += pe;
                Plds[wv][g * 8 + r][jt * 16 + ln] = (_Float16)pe;
            }
#pragma unroll
            for (int msk = 1; msk < 16; msk <<= 1)
                rowsum += __shfl_xor(rowsum, msk, 32);
            lr[r] = lr[r] * al[r] + rowsum;
            mr[r] = nm[r];
#pragma unroll
            for (int nt = 0; nt < 8; ++nt) o[nt][r] *= al[r];
        }

        // ---------------- O += P * V ---------------------------------------
#pragma unroll
        for (int kc = 0; kc < 2; ++kc) {
            v16h pa = ldA(&Plds[wv][ln][0], kc * 32, g);
#pragma unroll
            for (int nt = 0; nt < 8; ++nt) {
                v16h bf = ldB(&Vtlds[nt * 16 + ln][0], kc * 32, g);
                o[nt] = wmma_f16(pa, bf, o[nt]);
            }
        }
    }

    // ---------------- finalize & scatter to y16 [b,t,32,32,1024] -----------
#pragma unroll
    for (int r = 0; r < 8; ++r) {
        const float inv = 1.0f / lr[r];
        const int qg = wv * 16 + g * 8 + r;
        const int tt2 = qg >> 6, p2 = qg & 63;
        const int t2 = tpair * 2 + tt2;
        const int y2 = wr * 8 + (p2 >> 3), x2 = wc * 8 + (p2 & 7);
        _Float16* dst =
            y16 + ((((size_t)b * 8 + t2) * 32 + y2) * 32 + x2) * 1024 + h * 128 + ln;
#pragma unroll
        for (int nt = 0; nt < 8; ++nt)
            dst[nt * 16] = (_Float16)(o[nt][r] * inv);
    }
}

// ---------------------------------------------------------------------------
// Host-side orchestration
// ---------------------------------------------------------------------------
extern "C" void kernel_launch(void* const* d_in, const int* in_sizes, int n_in,
                              void* d_out, int out_size, void* d_ws, size_t ws_size,
                              hipStream_t stream)
{
    (void)in_sizes; (void)n_in; (void)out_size; (void)ws_size;

    const float* x  = (const float*)d_in[0];
    // d_in[1], d_in[2]: qmask/kvmask (unused by reference forward)
    const float* Wq = (const float*)d_in[3];
    const float* bq = (const float*)d_in[4];
    const float* Wk = (const float*)d_in[5];
    const float* bk = (const float*)d_in[6];
    const float* Wv = (const float*)d_in[7];
    const float* bv = (const float*)d_in[8];
    const float* Wp = (const float*)d_in[9];
    const float* bp = (const float*)d_in[10];
    const float* pw = (const float*)d_in[11];
    const float* pb = (const float*)d_in[12];
    float* out = (float*)d_out;

    const size_t NTOK = 16384;            // B*T*H*W
    const size_t DIMc = 1024;
    const size_t TOKB = NTOK * DIMc;      // 16,777,216 elems
    const size_t WB   = DIMc * DIMc;      // 1,048,576 elems
    const size_t PTOK = 2 * 8 * 64;       // pooled tokens = 1024
    const size_t PB   = PTOK * DIMc;      // 1,048,576 elems

    char* ws = (char*)d_ws;
    size_t off = 0;
    _Float16* x16  = (_Float16*)(ws + off); off += TOKB * 2;
    _Float16* q16  = (_Float16*)(ws + off); off += TOKB * 2;
    _Float16* k16  = (_Float16*)(ws + off); off += TOKB * 2;
    _Float16* v16  = (_Float16*)(ws + off); off += TOKB * 2;
    _Float16* y16  = (_Float16*)(ws + off); off += TOKB * 2;
    _Float16* wq16 = (_Float16*)(ws + off); off += WB * 2;
    _Float16* wk16 = (_Float16*)(ws + off); off += WB * 2;
    _Float16* wv16 = (_Float16*)(ws + off); off += WB * 2;
    _Float16* wp16 = (_Float16*)(ws + off); off += WB * 2;
    _Float16* px16 = (_Float16*)(ws + off); off += PB * 2;
    _Float16* pk16 = (_Float16*)(ws + off); off += PB * 2;
    _Float16* pv16 = (_Float16*)(ws + off); off += PB * 2;

    // 1) conversions to f16
    cvt_f32_f16<<<(int)((TOKB + 255) / 256), 256, 0, stream>>>(x, x16, (int)TOKB);
    cvt_f32_f16<<<(int)((WB + 255) / 256), 256, 0, stream>>>(Wq, wq16, (int)WB);
    cvt_f32_f16<<<(int)((WB + 255) / 256), 256, 0, stream>>>(Wk, wk16, (int)WB);
    cvt_f32_f16<<<(int)((WB + 255) / 256), 256, 0, stream>>>(Wv, wv16, (int)WB);
    cvt_f32_f16<<<(int)((WB + 255) / 256), 256, 0, stream>>>(Wp, wp16, (int)WB);

    // 2) Q/K/V projections: [16384,1024] x [1024,1024]^T
    gemm_nt<_Float16><<<dim3(128, 8), 256, 0, stream>>>(x16, wq16, bq, q16, 16384, 1024, 1024);
    gemm_nt<_Float16><<<dim3(128, 8), 256, 0, stream>>>(x16, wk16, bk, k16, 16384, 1024, 1024);
    gemm_nt<_Float16><<<dim3(128, 8), 256, 0, stream>>>(x16, wv16, bv, v16, 16384, 1024, 1024);

    // 3) depthwise 4x4 pool -> px16 [2,8,8,8,1024]
    pool_kernel<<<4096, 256, 0, stream>>>(x, pw, pb, px16);

    // 4) pooled K/V projections: [1024,1024] x [1024,1024]^T
    gemm_nt<_Float16><<<dim3(8, 8), 256, 0, stream>>>(px16, wk16, bk, pk16, 1024, 1024, 1024);
    gemm_nt<_Float16><<<dim3(8, 8), 256, 0, stream>>>(px16, wv16, bv, pv16, 1024, 1024, 1024);

    // 5) fused attention: 2*64*8 = 1024 problems
    attn_kernel<<<1024, 256, 0, stream>>>(q16, k16, v16, pk16, pv16, y16);

    // 6) output projection -> f32 d_out
    gemm_nt<float><<<dim3(128, 8), 256, 0, stream>>>(y16, wp16, bp, out, 16384, 1024, 1024);
}
